// Attention_9165460209905
// MI455X (gfx1250) — matrix operand order repaired and checked by reading, not compiled, not run on hardware
//
#include <hip/hip_runtime.h>
#include <hip/hip_bf16.h>

typedef __attribute__((ext_vector_type(16))) _Float16 v16h;
typedef __attribute__((ext_vector_type(8)))  float    v8f;

#define DIM    1024
#define HEADS  16
#define DHEAD  64
#define SEQ    2048
#define BATCH  2
#define SCALE  0.125f            // 64^-0.5
#define NEGINF (-3.0e38f)

// 48-byte fragment slot: v16h payload + 16B pad (bank-conflict-friendly 12-dword
// stride; slot and both 16B halves stay 16B aligned for b128 / async copies).
struct __align__(16) Frag {
  v16h  v;
  uint4 pad;
};
#define SLOT_H 24   // halves per slot (16 payload + 8 pad) in packed GLOBAL layouts

#define WMMA_F16(a, b, c) \
  __builtin_amdgcn_wmma_f32_16x16x32_f16(false, (a), false, (b), (short)0, (c), false, false)

// CDNA5 16x16x32 f16 fragment maps (ISA 7.12.2, wave32):
//  A 16x32 (MxK): lane = M + 16*bit3(K), half idx i = (K&7) | (bit4(K)<<3)
//  B 32x16 (KxN): lane = N + 16*bit3(K), half idx i = (K&7) | (bit4(K)<<3)

// --- CDNA5 async copy: 16B global -> 16B LDS (per-lane LDS dest), ASYNCcnt ---
__device__ __forceinline__ void cp_async16(unsigned lds_off, const _Float16* g) {
  asm volatile("global_load_async_to_lds_b128 %0, %1, off"
               :: "v"(lds_off), "v"(g) : "memory");
}
template <int N>
__device__ __forceinline__ void async_wait_le() {
  asm volatile("s_wait_asynccnt %0" :: "n"(N) : "memory");
}
__device__ __forceinline__ void ds_wait0() {
  asm volatile("s_wait_dscnt 0x0" ::: "memory");
}
__device__ __forceinline__ unsigned lds_off(const void* p) {
  return (unsigned)(uintptr_t)p;   // generic LDS ptr low 32 bits = LDS offset
}

// ---------------------------------------------------------------------------
// Kernel 1a: fp32 -> fp16 (x, row-major)
// ---------------------------------------------------------------------------
__global__ void cvt_f32_to_f16(const float* __restrict__ in,
                               _Float16* __restrict__ out, int n) {
  for (int i = blockIdx.x * blockDim.x + threadIdx.x; i < n;
       i += gridDim.x * blockDim.x)
    out[i] = (_Float16)in[i];
}

// ---------------------------------------------------------------------------
// Kernel 1b: pack weights [K=1024][N] f32 -> fragment-major f16 so B-tile
// staging is a LINEAR stream of 16B async copies.
// ---------------------------------------------------------------------------
__global__ void pack_w(const float* __restrict__ w, _Float16* __restrict__ wp, int N) {
  int total = DIM * N;
  for (int idx = blockIdx.x * blockDim.x + threadIdx.x; idx < total;
       idx += gridDim.x * blockDim.x) {
    int k = idx / N, n = idx % N;
    size_t slot = ((((size_t)(n >> 6) * (DIM / 32) + (k >> 5)) * 4 + ((n >> 4) & 3)) * 32)
                  + (n & 15) + ((k & 8) << 1);
    wp[slot * SLOT_H + ((k & 7) | ((k & 16) >> 1))] = (_Float16)w[idx];
  }
}

// ---------------------------------------------------------------------------
// Kernel 2: QKV projection, 128x64 tile / 128-thread block, double-buffered
// async pipeline.  Q -> qh (pre-scaled), K -> kh, V -> vp (fragment-major).
// ---------------------------------------------------------------------------
__global__ __launch_bounds__(128) void qkv_gemm(const _Float16* __restrict__ X,   // [4096,1024]
                                                const _Float16* __restrict__ Wp,  // packed
                                                _Float16* __restrict__ qh,
                                                _Float16* __restrict__ kh,
                                                _Float16* __restrict__ vp) {
  __shared__ Frag AsF[2][8][32];
  __shared__ Frag BsF[2][4][32];
  const int wave = threadIdx.x >> 5;
  const int lane = threadIdx.x & 31;
  const int row0 = blockIdx.x * 128;
  const int col0 = blockIdx.y * 64;

  auto stageAB = [&](int ks, int buf) {
    const int k0 = ks * 32;
#pragma unroll
    for (int c = 0; c < 4; ++c) {                 // A (128x32): 16B async scatter
      int ch = threadIdx.x + 128 * c;
      int r  = ch >> 2;
      int c8 = (ch & 3) << 3;
      const _Float16* src = X + (size_t)(row0 + r) * DIM + k0 + c8;
      _Float16* dst = (_Float16*)&AsF[buf][r >> 4][(r & 15) + (((c8 >> 3) & 1) << 4)]
                      + (((c8 >> 4) & 1) << 3);
      cp_async16(lds_off(dst), src);
    }
    const _Float16* wsrc = Wp + (((size_t)(col0 >> 6) * (DIM / 32) + ks) * 128) * SLOT_H;
    const unsigned bsf0 = lds_off(&BsF[buf][0][0]);
#pragma unroll
    for (int c = 0; c < 2; ++c) {                 // B: linear async stream
      int ch   = threadIdx.x + 128 * c;
      int slot = ch >> 1, hh = ch & 1;
      cp_async16(bsf0 + slot * 48 + hh * 16, wsrc + slot * SLOT_H + hh * 8);
    }
  };

  v8f acc[2][4] = {};
  const int NK = DIM / 32;

  stageAB(0, 0);
  for (int ks = 0; ks < NK; ++ks) {
    const int buf = ks & 1;
    if (ks + 1 < NK) {
      stageAB(ks + 1, buf ^ 1);    // prefetch next tile behind the WMMAs
      async_wait_le<6>();          // in-order: current tile's 6 copies done
    } else {
      async_wait_le<0>();
    }
    __syncthreads();

    v16h a0 = AsF[buf][wave * 2 + 0][lane].v;
    v16h a1 = AsF[buf][wave * 2 + 1][lane].v;
#pragma unroll
    for (int t = 0; t < 4; ++t) {
      v16h b = BsF[buf][t][lane].v;
      acc[0][t] = WMMA_F16(a0, b, acc[0][t]);
      acc[1][t] = WMMA_F16(a1, b, acc[1][t]);
    }
    __syncthreads();               // frees this buffer for the next prefetch
  }

  const int n15 = lane & 15;
  const int mo  = (lane < 16) ? 0 : 8;
  const int which = col0 >> 10;
  const int head  = (col0 & 1023) >> 6;

  if (which == 2) {
    // V -> fragment-major: slot((keyblk,t,khalf,lane)) per (b,h)
#pragma unroll
    for (int u = 0; u < 2; ++u)
#pragma unroll
      for (int t = 0; t < 4; ++t)
#pragma unroll
        for (int r = 0; r < 8; ++r) {
          int mg   = row0 + (wave * 2 + u) * 16 + mo + r;
          int bidx = mg >> 11, key = mg & 2047;
          int slot = ((((key >> 6) * 4 + t) * 2) + ((key >> 5) & 1)) * 32
                     + n15 + ((key & 8) << 1);
          size_t off = ((size_t)(bidx * HEADS + head) * 8192 + slot) * SLOT_H + ((key & 7) | ((key & 16) >> 1));
          vp[off] = (_Float16)acc[u][t][r];
        }
  } else {
    _Float16* dst = (which == 0) ? qh : kh;
    float sc = (which == 0) ? SCALE : 1.0f;
#pragma unroll
    for (int u = 0; u < 2; ++u)
#pragma unroll
      for (int t = 0; t < 4; ++t) {
        int dd = t * 16 + n15;
#pragma unroll
        for (int r = 0; r < 8; ++r) {
          int mg   = row0 + (wave * 2 + u) * 16 + mo + r;
          int bidx = mg >> 11, nidx = mg & 2047;
          dst[((size_t)(bidx * HEADS + head) * SEQ + nidx) * DHEAD + dd] =
              (_Float16)(acc[u][t][r] * sc);
        }
      }
  }
}

// ---------------------------------------------------------------------------
// Kernel 3: causal flash attention, double-buffered K/V async pipeline.
// Q staging aliases K/V buffer 1 (consumed into registers before first use).
// ---------------------------------------------------------------------------
__global__ __launch_bounds__(128) void flash_attn(const _Float16* __restrict__ qh,
                                                  const _Float16* __restrict__ kh,
                                                  const _Float16* __restrict__ vp,
                                                  _Float16* __restrict__ O) {
  __shared__ Frag KVbuf[2][2][4][2][32];  // [buf][0=K,1=V][tile][k-half][lane]
  __shared__ Frag PsF[4][2][32];          // per-wave P staging (A frags)

  const int it = blockIdx.x;
  const int h  = blockIdx.y;
  const int b  = blockIdx.z;
  const int wave = threadIdx.x >> 5;
  const int lane = threadIdx.x & 31;
  const int row0 = it * 64;
  const int n15  = lane & 15;
  const int mo   = (lane < 16) ? 0 : 8;

  const _Float16* Q  = qh + (size_t)(b * HEADS + h) * SEQ * DHEAD;
  const _Float16* K  = kh + (size_t)(b * HEADS + h) * SEQ * DHEAD;
  const _Float16* Vp = vp + (size_t)(b * HEADS + h) * 8192 * SLOT_H;

  // --- Q (64x64) -> A-frag layout in KVbuf[1][0] (aliased) ---
  {
    Frag (*QsF)[2][32] = KVbuf[1][0];
#pragma unroll
    for (int c = 0; c < 4; ++c) {
      int ch = threadIdx.x + 128 * c;
      int r  = ch >> 3;
      int c8 = (ch & 7) << 3;
      const _Float16* src = Q + (size_t)(row0 + r) * DHEAD + c8;
      int cc = c8 & 31;
      _Float16* dst = (_Float16*)&QsF[r >> 4][c8 >> 5][(r & 15) + (((cc >> 3) & 1) << 4)]
                      + (((cc >> 4) & 1) << 3);
      cp_async16(lds_off(dst), src);
    }
  }
  async_wait_le<0>();
  __syncthreads();
  const v16h qa0 = KVbuf[1][0][wave][0][lane].v;
  const v16h qa1 = KVbuf[1][0][wave][1][lane].v;
  ds_wait0();                       // Q frags in registers before buf1 reuse
  __syncthreads();

  auto stageKV = [&](int j, int buf) {
#pragma unroll
    for (int c = 0; c < 4; ++c) {   // K: row-major == K^T B-frag chunks
      int ch  = threadIdx.x + 128 * c;
      int key = ch >> 3;
      int d8  = (ch & 7) << 3;
      const _Float16* src = K + (size_t)(j * 64 + key) * DHEAD + d8;
      _Float16* dst = (_Float16*)&KVbuf[buf][0][key >> 4][d8 >> 5]
                                      [(key & 15) + (((d8 >> 3) & 1) << 4)] + (((d8 >> 4) & 1) << 3);
      cp_async16(lds_off(dst), src);
    }
    const _Float16* vsrc = Vp + (size_t)j * 256 * SLOT_H;
    const unsigned vsf0 = lds_off(&KVbuf[buf][1][0][0][0]);
#pragma unroll
    for (int c = 0; c < 4; ++c) {   // V: linear async stream
      int ch   = threadIdx.x + 128 * c;
      int slot = ch >> 1, hh = ch & 1;
      cp_async16(vsf0 + slot * 48 + hh * 16, vsrc + slot * SLOT_H + hh * 8);
    }
  };

  float mrow[8], lrow[8];
#pragma unroll
  for (int r = 0; r < 8; ++r) { mrow[r] = NEGINF; lrow[r] = 0.0f; }
  v8f oacc[4] = {};
  const int laneHi = ((n15 >> 3) & 1) << 4;

  stageKV(0, 0);
  for (int j = 0; j <= it; ++j) {
    const int  buf  = j & 1;
    const bool diag = (j == it);
    if (!diag) {
      stageKV(j + 1, buf ^ 1);      // prefetch next K/V behind compute
      async_wait_le<8>();
    } else {
      async_wait_le<0>();
    }
    __syncthreads();

    // --- S = Q * K^T : 8 WMMAs per wave ---
    v8f s[4];
#pragma unroll
    for (int t = 0; t < 4; ++t) {
      v8f z = {};
      z    = WMMA_F16(qa0, KVbuf[buf][0][t][0][lane].v, z);
      s[t] = WMMA_F16(qa1, KVbuf[buf][0][t][1][lane].v, z);
    }

    // --- (diagonal-only causal mask) + online softmax ---
#pragma unroll
    for (int r = 0; r < 8; ++r) {
      int rg = row0 + wave * 16 + mo + r;
      float mx = NEGINF;
      if (diag) {
#pragma unroll
        for (int t = 0; t < 4; ++t) {
          int cg = j * 64 + t * 16 + n15;
          if (cg > rg) s[t][r] = NEGINF;
          mx = fmaxf(mx, s[t][r]);
        }
      } else {
#pragma unroll
        for (int t = 0; t < 4; ++t) mx = fmaxf(mx, s[t][r]);
      }
#pragma unroll
      for (int m = 1; m < 16; m <<= 1) mx = fmaxf(mx, __shfl_xor(mx, m, 32));
      float mnew  = fmaxf(mrow[r], mx);
      float alpha = __expf(mrow[r] - mnew);
      float rsum  = 0.0f;
#pragma unroll
      for (int t = 0; t < 4; ++t) {
        float p = __expf(s[t][r] - mnew);
        s[t][r] = p;
        rsum += p;
      }
#pragma unroll
      for (int m = 1; m < 16; m <<= 1) rsum += __shfl_xor(rsum, m, 32);
      lrow[r] = lrow[r] * alpha + rsum;
      mrow[r] = mnew;
#pragma unroll
      for (int t = 0; t < 4; ++t) oacc[t][r] *= alpha;
    }

    // --- spill P into A-frag layout (per-wave buffer; intra-wave in-order) ---
#pragma unroll
    for (int t = 0; t < 4; ++t) {
      int khalf = t >> 1;
      int i     = (n15 & 7) | ((t & 1) << 3);
#pragma unroll
      for (int r = 0; r < 8; ++r)
        ((_Float16*)&PsF[wave][khalf][mo + r + laneHi])[i] = (_Float16)s[t][r];
    }
    v16h pa0 = PsF[wave][0][lane].v;
    v16h pa1 = PsF[wave][1][lane].v;

    // --- O += P * V : 8 WMMAs per wave ---
#pragma unroll
    for (int t = 0; t < 4; ++t) {
      oacc[t] = WMMA_F16(pa0, KVbuf[buf][1][t][0][lane].v, oacc[t]);
      oacc[t] = WMMA_F16(pa1, KVbuf[buf][1][t][1][lane].v, oacc[t]);
    }
    __syncthreads();               // frees this buffer for the next prefetch
  }

  // --- normalize + store O: [b][n][h*64+d] ---
#pragma unroll
  for (int r = 0; r < 8; ++r) {
    int rg = row0 + wave * 16 + mo + r;
    float inv = 1.0f / lrow[r];
#pragma unroll
    for (int t = 0; t < 4; ++t)
      O[((size_t)b * SEQ + rg) * (HEADS * DHEAD) + h * DHEAD + t * 16 + n15] =
          (_Float16)(oacc[t][r] * inv);
  }
}

// ---------------------------------------------------------------------------
// Kernel 4: output projection, same double-buffered pipeline, f32 out + bias
// ---------------------------------------------------------------------------
__global__ __launch_bounds__(128) void out_gemm(const _Float16* __restrict__ A,   // [4096,1024]
                                                const _Float16* __restrict__ Wp,  // packed
                                                const float* __restrict__ bias,
                                                float* __restrict__ out) {
  __shared__ Frag AsF[2][8][32];
  __shared__ Frag BsF[2][4][32];
  const int wave = threadIdx.x >> 5;
  const int lane = threadIdx.x & 31;
  const int row0 = blockIdx.x * 128;
  const int col0 = blockIdx.y * 64;

  auto stageAB = [&](int ks, int buf) {
    const int k0 = ks * 32;
#pragma unroll
    for (int c = 0; c < 4; ++c) {
      int ch = threadIdx.x + 128 * c;
      int r  = ch >> 2;
      int c8 = (ch & 3) << 3;
      const _Float16* src = A + (size_t)(row0 + r) * DIM + k0 + c8;
      _Float16* dst = (_Float16*)&AsF[buf][r >> 4][(r & 15) + (((c8 >> 3) & 1) << 4)]
                      + (((c8 >> 4) & 1) << 3);
      cp_async16(lds_off(dst), src);
    }
    const _Float16* wsrc = Wp + (((size_t)(col0 >> 6) * (DIM / 32) + ks) * 128) * SLOT_H;
    const unsigned bsf0 = lds_off(&BsF[buf][0][0]);
#pragma unroll
    for (int c = 0; c < 2; ++c) {
      int ch   = threadIdx.x + 128 * c;
      int slot = ch >> 1, hh = ch & 1;
      cp_async16(bsf0 + slot * 48 + hh * 16, wsrc + slot * SLOT_H + hh * 8);
    }
  };

  v8f acc[2][4] = {};
  const int NK = DIM / 32;

  stageAB(0, 0);
  for (int ks = 0; ks < NK; ++ks) {
    const int buf = ks & 1;
    if (ks + 1 < NK) {
      stageAB(ks + 1, buf ^ 1);
      async_wait_le<6>();
    } else {
      async_wait_le<0>();
    }
    __syncthreads();

    v16h a0 = AsF[buf][wave * 2 + 0][lane].v;
    v16h a1 = AsF[buf][wave * 2 + 1][lane].v;
#pragma unroll
    for (int t = 0; t < 4; ++t) {
      v16h b = BsF[buf][t][lane].v;
      acc[0][t] = WMMA_F16(a0, b, acc[0][t]);
      acc[1][t] = WMMA_F16(a1, b, acc[1][t]);
    }
    __syncthreads();
  }

  const int n15 = lane & 15;
  const int mo  = (lane < 16) ? 0 : 8;
#pragma unroll
  for (int u = 0; u < 2; ++u)
#pragma unroll
    for (int t = 0; t < 4; ++t) {
      int cg   = col0 + t * 16 + n15;
      float bb = bias[cg];
#pragma unroll
      for (int r = 0; r < 8; ++r) {
        int mg = row0 + (wave * 2 + u) * 16 + mo + r;
        out[(size_t)mg * DIM + cg] = acc[u][t][r] + bb;
      }
    }
}

// ---------------------------------------------------------------------------
// Host-side pipeline
// ---------------------------------------------------------------------------
extern "C" void kernel_launch(void* const* d_in, const int* in_sizes, int n_in,
                              void* d_out, int out_size, void* d_ws, size_t ws_size,
                              hipStream_t stream) {
  const float* x     = (const float*)d_in[0];   // [2,2048,1024]
  const float* w_qkv = (const float*)d_in[1];   // [1024,3072]
  const float* w_out = (const float*)d_in[2];   // [1024,1024]
  const float* b_out = (const float*)d_in[3];   // [1024]
  float* out = (float*)d_out;                   // [2,2048,1024]

  const int    NX     = BATCH * SEQ * DIM;                       // 4,194,304
  const size_t NWQKVP = (size_t)48 * 32 * 128 * SLOT_H;          // 4,718,592
  const size_t NWOUTP = (size_t)16 * 32 * 128 * SLOT_H;          // 1,572,864
  const size_t NQ     = (size_t)BATCH * HEADS * SEQ * DHEAD;     // 4,194,304
  const size_t NVP    = (size_t)BATCH * HEADS * 8192 * SLOT_H;   // 6,291,456

  _Float16* xh    = (_Float16*)d_ws;
  _Float16* wqkvp = xh + NX;
  _Float16* woutp = wqkvp + NWQKVP;
  _Float16* qh    = woutp + NWOUTP;
  _Float16* kh    = qh + NQ;
  _Float16* vp    = kh + NQ;
  _Float16* oh    = vp + NVP;
  (void)ws_size; (void)in_sizes; (void)n_in; (void)out_size;

  cvt_f32_to_f16<<<(NX + 255) / 256, 256, 0, stream>>>(x, xh, NX);
  pack_w<<<4096, 256, 0, stream>>>(w_qkv, wqkvp, 3 * DIM);
  pack_w<<<2048, 256, 0, stream>>>(w_out, woutp, DIM);

  {
    dim3 grid((BATCH * SEQ) / 128, (3 * DIM) / 64);   // 32 x 48
    qkv_gemm<<<grid, 128, 0, stream>>>(xh, wqkvp, qh, kh, vp);
  }
  {
    dim3 grid(SEQ / 64, HEADS, BATCH);                // 32 x 16 x 2
    flash_attn<<<grid, 128, 0, stream>>>(qh, kh, vp, oh);
  }
  {
    dim3 grid((BATCH * SEQ) / 128, DIM / 64);         // 32 x 16
    out_gemm<<<grid, 128, 0, stream>>>(oh, woutp, b_out, out);
  }
}
